// TransMILGraphAggregator_56014963475229
// MI455X (gfx1250) — compile-verified, matrix operand order checked by reading, not run on hardware
//
#include <hip/hip_runtime.h>
#include <hip/hip_bf16.h>
#include <math.h>

// ---------------- types ----------------
typedef __attribute__((ext_vector_type(16))) __bf16 v16bf;
typedef __attribute__((ext_vector_type(8)))  __bf16 v8bf;
typedef __attribute__((ext_vector_type(8)))  float  v8f;
typedef __attribute__((ext_vector_type(2)))  float  v2f;
typedef __attribute__((ext_vector_type(4)))  unsigned int u32x4;
typedef __attribute__((ext_vector_type(8)))  unsigned int u32x8;

#define BB 4
#define NN 4096
#define DD 768
#define MTOT (BB*NN)        // 16384
#define KTOP 8

// ---------------------------------------------------------------
// Kernel 0: zero the output accumulator (d_out is poisoned by harness)
// ---------------------------------------------------------------
__global__ void zero_kernel(float* out, int n) {
    int i = blockIdx.x * blockDim.x + threadIdx.x;
    if (i < n) out[i] = 0.0f;
}

// ---------------------------------------------------------------
// Kernel 1: projections  Q = feats@Wq + bq ; [K|V] = feats@Wkv + bkv
// f32 WMMA 16x16x4. One wave computes a 64(M) x 64(N) tile.
// ---------------------------------------------------------------
__global__ __launch_bounds__(32)
void proj_kernel(const float* __restrict__ feats,
                 const float* __restrict__ Wq,  const float* __restrict__ bq,
                 const float* __restrict__ Wkv, const float* __restrict__ bkv,
                 float* __restrict__ Q, float* __restrict__ Kf, float* __restrict__ Vf) {
    const int lane   = threadIdx.x & 31;
    const int lane16 = lane & 15;
    const int hi     = lane >> 4;
    const int m0 = blockIdx.x * 64;          // row block
    const int c0 = blockIdx.y * 64;          // logical column block in [0,2304)

    const float* W;  const float* bias;  float* outp;
    int wc0, ldw, oc0;
    if (c0 < DD)            { W = Wq;  bias = bq;  outp = Q;  wc0 = c0;        ldw = DD;     oc0 = c0; }
    else if (c0 < 2*DD)     { W = Wkv; bias = bkv; outp = Kf; wc0 = c0 - DD;   ldw = 2*DD;   oc0 = c0 - DD; }
    else                    { W = Wkv; bias = bkv; outp = Vf; wc0 = c0 - DD;   ldw = 2*DD;   oc0 = c0 - 2*DD; }

    v8f acc[4][4];
    #pragma unroll
    for (int mt = 0; mt < 4; ++mt)
        #pragma unroll
        for (int t = 0; t < 4; ++t)
            #pragma unroll
            for (int j = 0; j < 8; ++j) acc[mt][t][j] = 0.0f;

    #pragma unroll 2
    for (int k = 0; k < DD; k += 4) {
        const int kk = k + hi * 2;
        v2f a[4], bop[4];
        #pragma unroll
        for (int mt = 0; mt < 4; ++mt) {
            const float* ap = feats + (size_t)(m0 + mt * 16 + lane16) * DD + kk;
            a[mt][0] = ap[0];
            a[mt][1] = ap[1];
        }
        #pragma unroll
        for (int t = 0; t < 4; ++t) {
            const int col = wc0 + t * 16 + lane16;
            bop[t][0] = W[(size_t)kk * ldw + col];
            bop[t][1] = W[(size_t)(kk + 1) * ldw + col];
        }
        #pragma unroll
        for (int mt = 0; mt < 4; ++mt)
            #pragma unroll
            for (int t = 0; t < 4; ++t)
                acc[mt][t] = __builtin_amdgcn_wmma_f32_16x16x4_f32(
                    false, a[mt], false, bop[t], (short)0, acc[mt][t], false, false);
    }

    float bv[4];
    #pragma unroll
    for (int t = 0; t < 4; ++t) bv[t] = bias[wc0 + t * 16 + lane16];

    #pragma unroll
    for (int mt = 0; mt < 4; ++mt)
        #pragma unroll
        for (int t = 0; t < 4; ++t)
            #pragma unroll
            for (int j = 0; j < 8; ++j) {
                const int r = m0 + mt * 16 + j + hi * 8;
                outp[(size_t)r * DD + oc0 + t * 16 + lane16] = acc[mt][t][j] + bv[t];
            }
}

// ---------------------------------------------------------------
// Kernel 2: L2-normalize rows of Q and K into bf16 copies
// ---------------------------------------------------------------
__global__ __launch_bounds__(256)
void norm_kernel(const float* __restrict__ Q, const float* __restrict__ Kf,
                 __bf16* __restrict__ Qn, __bf16* __restrict__ Kn) {
    const int row = blockIdx.x;              // 0..2*MTOT-1
    const float* src;  __bf16* dst;
    if (row < MTOT) { src = Q  + (size_t)row * DD;          dst = Qn + (size_t)row * DD; }
    else            { src = Kf + (size_t)(row - MTOT) * DD; dst = Kn + (size_t)(row - MTOT) * DD; }

    const int t = threadIdx.x;
    const float x0 = src[t], x1 = src[t + 256], x2 = src[t + 512];
    __shared__ float red[256];
    red[t] = x0 * x0 + x1 * x1 + x2 * x2;
    __syncthreads();
    #pragma unroll
    for (int off = 128; off > 0; off >>= 1) {
        if (t < off) red[t] += red[t + off];
        __syncthreads();
    }
    const float invn = 1.0f / fmaxf(sqrtf(red[0]), 1e-12f);
    dst[t]       = (__bf16)(x0 * invn);
    dst[t + 256] = (__bf16)(x1 * invn);
    dst[t + 512] = (__bf16)(x2 * invn);
}

// ---------------------------------------------------------------
// TDM: 2D tile load (rows x tile_d0_dw dwords, row stride stride_dw dwords)
// from global to LDS. Descriptor groups 0/1 only (<=2D tensor).
// ---------------------------------------------------------------
__device__ __forceinline__ unsigned lds_offset(const void* p) {
    // generic LDS address: low 32 bits are the LDS byte offset
    return (unsigned)(unsigned long long)p;
}

__device__ __forceinline__ void tdm_load_2d(unsigned lds_off, const void* gptr,
                                            unsigned tile_d0_dw, unsigned rows,
                                            unsigned stride_dw) {
    const unsigned long long ga = (unsigned long long)gptr;
    u32x4 g0;
    g0[0] = 1u;                                                  // count=1 (valid)
    g0[1] = lds_off;                                             // lds_addr
    g0[2] = (unsigned)ga;                                        // global_addr[31:0]
    g0[3] = ((unsigned)(ga >> 32) & 0x01FFFFFFu) | (2u << 30);   // addr[56:32] | type=2
    u32x8 g1;
    g1[0] = (2u << 16);                                          // data_size = 4B
    g1[1] = (tile_d0_dw & 0xFFFFu) << 16;                        // tensor_dim0[15:0]
    g1[2] = ((tile_d0_dw >> 16) & 0xFFFFu) | ((rows & 0xFFFFu) << 16); // td0 hi | td1 lo
    g1[3] = ((rows >> 16) & 0xFFFFu) | ((tile_d0_dw & 0xFFFFu) << 16); // td1 hi | tile_dim0
    g1[4] = rows & 0xFFFFu;                                      // tile_dim1 (tile_dim2=0)
    g1[5] = stride_dw;                                           // tensor_dim0_stride lo32
    g1[6] = 0u;
    g1[7] = 0u;
    asm volatile("tensor_load_to_lds %0, %1" :: "s"(g0), "s"(g1) : "memory");
}

// ---------------------------------------------------------------
// Kernel 3: fused cosine-sim + top-8 (bf16 WMMA 16x16x32)
// Block: 8 waves = 128 queries of one batch. Candidate K-rows stream
// through a double-buffered LDS panel (16 rows x 768 bf16 = 24KB) filled
// by the Tensor Data Mover; all 8 waves reuse each panel (8x traffic cut).
// A = candidates (M), B = queries (N) -> each lane owns 8 candidate sims
// for ONE query; lane pair (l, l+16) merged at the end via shuffles.
// ---------------------------------------------------------------
__device__ __forceinline__ v16bf load_row32(const __bf16* rowp, int kc, int hi) {
    // per-lane 16-bit operand layout: halves K = {0..7,16..23} (lanes<16)
    //                                 or     K = {8..15,24..31} (lanes>=16)
    union { v16bf v; v8bf h[2]; } u;
    const __bf16* p = rowp + kc * 32 + hi * 8;
    u.h[0] = *(const v8bf*)(p);
    u.h[1] = *(const v8bf*)(p + 16);
    return u.v;
}

__device__ __forceinline__ void topk_insert(float (&tv)[KTOP], int (&ti)[KTOP], float v, int c) {
    if (v <= tv[KTOP - 1]) return;
    float cv = v; int ci = c;
    #pragma unroll
    for (int s = 0; s < KTOP; ++s) {
        if (cv > tv[s]) {
            float t0 = tv[s]; int t1 = ti[s];
            tv[s] = cv; ti[s] = ci;
            cv = t0; ci = t1;
        }
    }
}

#define NPANEL (NN / 16)    // 256 candidate panels of 16 rows

__global__ __launch_bounds__(256)
void simtopk_kernel(const __bf16* __restrict__ Qn, const __bf16* __restrict__ Kn,
                    int* __restrict__ idxout) {
    __shared__ __align__(16) __bf16 panel[2][16 * DD];   // 2 x 24KB double buffer

    const int b   = blockIdx.y;
    const int tid = threadIdx.x, wave = tid >> 5, lane = tid & 31;
    const int lane16 = lane & 15, hi = lane >> 4;
    const int q0w = blockIdx.x * 128 + wave * 16;        // this wave's query strip

    const __bf16* Qb = Qn + (size_t)(b * NN + q0w + lane16) * DD;
    const char*   Kbase = (const char*)(Kn + (size_t)b * NN * DD);

    v16bf qreg[24];
    #pragma unroll
    for (int kc = 0; kc < 24; ++kc) qreg[kc] = load_row32(Qb, kc, hi);

    float tv[KTOP]; int ti[KTOP];
    #pragma unroll
    for (int s = 0; s < KTOP; ++s) { tv[s] = -1e30f; ti[s] = 0; }
    const int myq = q0w + lane16;

    // prologue: DMA panel 0 into buffer 0
    if (wave == 0) {
        tdm_load_2d(lds_offset(&panel[0][0]), Kbase, DD / 2, 16, DD / 2);
        __builtin_amdgcn_s_wait_tensorcnt(0);
    }
    __syncthreads();

    for (int cb = 0; cb < NPANEL; ++cb) {
        const int buf = cb & 1;
        if (wave == 0 && cb + 1 < NPANEL) {
            tdm_load_2d(lds_offset(&panel[buf ^ 1][0]),
                        Kbase + (size_t)(cb + 1) * 16 * DD * sizeof(__bf16),
                        DD / 2, 16, DD / 2);
        }

        // compute on panel[buf]: 16 candidates x 16 queries, K = 768
        v8f acc;
        #pragma unroll
        for (int j = 0; j < 8; ++j) acc[j] = 0.0f;
        const __bf16* rp = &panel[buf][lane16 * DD];
        #pragma unroll
        for (int kc = 0; kc < 24; ++kc) {
            v16bf aop = load_row32(rp, kc, hi);
            acc = __builtin_amdgcn_wmma_f32_16x16x32_bf16(
                false, aop, false, qreg[kc], (short)0, acc, false, false);
        }
        const int c0 = cb * 16;
        #pragma unroll
        for (int j = 0; j < 8; ++j) {
            const int cand = c0 + hi * 8 + j;   // C layout: M = j (lanes<16) / j+8
            const float v = acc[j];
            if (cand != myq) topk_insert(tv, ti, v, cand);
        }

        if (wave == 0 && cb + 1 < NPANEL) __builtin_amdgcn_s_wait_tensorcnt(0);
        __syncthreads();
    }

    // merge lane pair (hi=0 half with hi=1 half) via shuffles
    float pv[KTOP]; int pi[KTOP];
    #pragma unroll
    for (int s = 0; s < KTOP; ++s) {
        pv[s] = __shfl_xor(tv[s], 16, 32);
        pi[s] = __shfl_xor(ti[s], 16, 32);
    }
    #pragma unroll
    for (int s = 0; s < KTOP; ++s) topk_insert(tv, ti, pv[s], pi[s]);

    if (hi == 0) {
        int* op = idxout + (size_t)(b * NN + q0w + lane16) * KTOP;
        #pragma unroll
        for (int s = 0; s < KTOP; ++s) op[s] = ti[s];
    }
}

// ---------------------------------------------------------------
// Kernel 4: gather neighbors, exact fp32 attention, mean-accumulate
// ---------------------------------------------------------------
__global__ __launch_bounds__(256)
void attn_kernel(const float* __restrict__ Q, const float* __restrict__ Kf,
                 const float* __restrict__ Vf, const int* __restrict__ idx,
                 float* __restrict__ out) {
    const int row = blockIdx.x;       // 0..MTOT-1
    const int b   = row >> 12;        // /NN
    const int tid = threadIdx.x, wave = tid >> 5, lane = tid & 31;

    const float* q  = Q + (size_t)row * DD;
    const int*   id = idx + (size_t)row * KTOP;

    __shared__ float sdot[KTOP];
    {
        const int nb = id[wave];
        const float* krow = Kf + ((size_t)b * NN + nb) * DD;
        float p = 0.0f;
        for (int d = lane; d < DD; d += 32) p += q[d] * krow[d];
        #pragma unroll
        for (int off = 16; off >= 1; off >>= 1) p += __shfl_xor(p, off, 32);
        if (lane == 0) sdot[wave] = p;
    }
    __syncthreads();

    const float scale = 0.03608439182435161f;   // 1/sqrt(768)
    float dj[KTOP];
    float m = -1e30f;
    #pragma unroll
    for (int j = 0; j < KTOP; ++j) { dj[j] = sdot[j] * scale; m = fmaxf(m, dj[j]); }
    float se = 0.0f;
    #pragma unroll
    for (int j = 0; j < KTOP; ++j) { dj[j] = __expf(dj[j] - m); se += dj[j]; }
    const float wscale = (1.0f / se) * (1.0f / (float)NN);

    for (int d = tid; d < DD; d += 256) {
        float o = 0.0f;
        #pragma unroll
        for (int j = 0; j < KTOP; ++j)
            o += dj[j] * Vf[((size_t)b * NN + id[j]) * DD + d];
        atomicAdd(&out[b * DD + d], o * wscale);
    }
}

// ---------------------------------------------------------------
// launch
// ---------------------------------------------------------------
extern "C" void kernel_launch(void* const* d_in, const int* in_sizes, int n_in,
                              void* d_out, int out_size, void* d_ws, size_t ws_size,
                              hipStream_t stream) {
    const float* feats = (const float*)d_in[0];   // [B,N,D]
    const float* Wq    = (const float*)d_in[1];   // [D,D]
    const float* bq    = (const float*)d_in[2];   // [D]
    const float* Wkv   = (const float*)d_in[3];   // [D,2D]
    const float* bkv   = (const float*)d_in[4];   // [2D]
    float* out = (float*)d_out;                   // [B,D]

    char* ws = (char*)d_ws;
    const size_t fsz = (size_t)MTOT * DD * sizeof(float);   // 50,331,648
    const size_t bsz = (size_t)MTOT * DD * sizeof(__bf16);  // 25,165,824
    float*  Q   = (float*)(ws);
    float*  Kf  = (float*)(ws + fsz);
    float*  Vf  = (float*)(ws + 2 * fsz);
    __bf16* Qn  = (__bf16*)(ws + 3 * fsz);
    __bf16* Kn  = (__bf16*)(ws + 3 * fsz + bsz);
    int*    idx = (int*)(ws + 3 * fsz + 2 * bsz);

    // 0: zero output
    zero_kernel<<<dim3((BB * DD + 255) / 256), dim3(256), 0, stream>>>(out, BB * DD);

    // 1: projections (f32 WMMA)
    proj_kernel<<<dim3(MTOT / 64, (3 * DD) / 64), dim3(32), 0, stream>>>(
        feats, Wq, bq, Wkv, bkv, Q, Kf, Vf);

    // 2: normalize -> bf16
    norm_kernel<<<dim3(2 * MTOT), dim3(256), 0, stream>>>(Q, Kf, Qn, Kn);

    // 3: fused sim + top-8 (bf16 WMMA, TDM-staged candidate panels)
    simtopk_kernel<<<dim3(NN / 128, BB), dim3(256), 0, stream>>>(Qn, Kn, idx);

    // 4: exact attention + mean
    attn_kernel<<<dim3(MTOT), dim3(256), 0, stream>>>(Q, Kf, Vf, idx, out);
}
